// TemporalCrossTransformer_7421703487805
// MI455X (gfx1250) — matrix operand order repaired and checked by reading, not compiled, not run on hardware
//
#include <hip/hip_runtime.h>
#include <hip/hip_bf16.h>
#include <math.h>

// ---------------------------------------------------------------------------
// TemporalCrossTransformer for MI455X (gfx1250), bf16 WMMA pipeline.
// Workspace requirement: ~95.1 MB (regions aliased over time, see offsets).
// ---------------------------------------------------------------------------

#define SEQ_LEN 8
#define D_IN    2048
#define D_OUT   1152
#define TSS     3
#define WAY     5
#define SHOT    5
#define N_SUP   25
#define N_Q     100
#define N_ALL   125           // 25 support + 100 query samples
#define T_LEN   56            // C(8,3)
#define D_FLAT  6144
#define KS      280           // SHOT * T_LEN (softmax width)
#define MROWS_F 1000          // N_ALL * SEQ_LEN   (feature GEMM M)
#define MROWS_Q 5600          // N_Q * T_LEN       (attention GEMM M)
#define LN_EPS  1e-5f
#define SM_SCALE 0.0294627825f   // 1/sqrt(1152)

// lexicographic combinations(range(8), 3)
__constant__ int TUP[T_LEN][3] = {
 {0,1,2},{0,1,3},{0,1,4},{0,1,5},{0,1,6},{0,1,7},
 {0,2,3},{0,2,4},{0,2,5},{0,2,6},{0,2,7},
 {0,3,4},{0,3,5},{0,3,6},{0,3,7},
 {0,4,5},{0,4,6},{0,4,7},
 {0,5,6},{0,5,7},
 {0,6,7},
 {1,2,3},{1,2,4},{1,2,5},{1,2,6},{1,2,7},
 {1,3,4},{1,3,5},{1,3,6},{1,3,7},
 {1,4,5},{1,4,6},{1,4,7},
 {1,5,6},{1,5,7},
 {1,6,7},
 {2,3,4},{2,3,5},{2,3,6},{2,3,7},
 {2,4,5},{2,4,6},{2,4,7},
 {2,5,6},{2,5,7},
 {2,6,7},
 {3,4,5},{3,4,6},{3,4,7},
 {3,5,6},{3,5,7},
 {3,6,7},
 {4,5,6},{4,5,7},
 {4,6,7},
 {5,6,7}
};

// ---------------- bf16 helpers / WMMA types --------------------------------
typedef __attribute__((ext_vector_type(16))) __bf16 v16bf;
typedef __attribute__((ext_vector_type(8)))  float  v8f;

union FragU { uint4 q[2]; v16bf v; };
union AccU  { v8f v; float f[8]; };
union U2    { uint2 u; unsigned short s[4]; };

__device__ __forceinline__ unsigned short f2bf(float x) {
  unsigned int u = __float_as_uint(x);
  unsigned int r = u + 0x7FFFu + ((u >> 16) & 1u);   // round-to-nearest-even
  return (unsigned short)(r >> 16);
}

__device__ __forceinline__ v8f wmma_bf16(v16bf a, v16bf b, v8f c) {
  return __builtin_amdgcn_wmma_f32_16x16x32_bf16(
      /*neg_a=*/false, a, /*neg_b=*/false, b,
      /*c_mod=*/(short)0, c, /*reuse_a=*/false, /*reuse_b=*/false);
}

// GEMM tile geometry: block = 256 threads = 8 waves; tile 128x64, K-step 32.
#define BM 128
#define BN 64
#define BK 32
#define LDP 40      // padded LDS row length (elements); 80B rows, 16B aligned

// A fragment (16x32 bf16): lane<16 -> K {0..7,16..23}, lane>=16 -> +8
__device__ __forceinline__ v16bf lds_frag_a(const unsigned short* lds, int rowBase, int lane) {
  int r  = rowBase + (lane & 15);
  int kb = (lane < 16) ? 0 : 8;
  const unsigned short* p = lds + r * LDP + kb;
  FragU f;
  f.q[0] = *(const uint4*)(p);
  f.q[1] = *(const uint4*)(p + 16);
  return f.v;
}

// B fragment (32x16 bf16), staged n-major in LDS: lane<16 -> K {0..7,16..23}, lane>=16 -> +8
__device__ __forceinline__ v16bf lds_frag_b(const unsigned short* lds, int colBase, int lane) {
  int n  = colBase + (lane & 15);
  int kb = (lane < 16) ? 0 : 8;
  const unsigned short* p = lds + n * LDP + kb;
  FragU f;
  f.q[0] = *(const uint4*)(p);
  f.q[1] = *(const uint4*)(p + 16);
  return f.v;
}

// ---------------- kernel 0a: x + positional encoding -> bf16 ---------------
__global__ __launch_bounds__(256) void prep_x_kernel(
    const float* __restrict__ sup, const float* __restrict__ qry,
    unsigned short* __restrict__ X) {
  size_t i = (size_t)blockIdx.x * 256 + threadIdx.x;
  const size_t total = (size_t)N_ALL * SEQ_LEN * D_IN;
  if (i >= total) return;
  int c = (int)(i % D_IN);
  size_t nr = i / D_IN;
  int r = (int)(nr % SEQ_LEN);
  size_t n = nr / SEQ_LEN;
  float x = (n < N_SUP) ? sup[i] : qry[i - (size_t)N_SUP * SEQ_LEN * D_IN];
  int p = c >> 1;
  // div = exp(-(2p) * ln(10000)/2048)
  float div = __expf(-(float)(2 * p) * 4.4972365e-3f);
  float ang = (float)r * div;
  float pe = (((c & 1) == 0) ? __sinf(ang) : __cosf(ang)) * 0.1f;
  X[i] = f2bf(x + pe);
}

// ---------------- kernel 0b: weights -> transposed bf16 --------------------
// Wt[z][n][k] = W[w][(z%3)*2048 + k][n],  z = w*3+j, n in [0,1152), k in [0,2048)
// Makes the feature GEMM an "NT" GEMM: B rows have contiguous K -> vector LDS
// staging instead of scalar ds_store_b16 transposes in the hot loop.
__global__ __launch_bounds__(256) void prep_wt_kernel(
    const float* __restrict__ kw, const float* __restrict__ vw,
    unsigned short* __restrict__ Wt) {
  __shared__ unsigned short tile[32][33];
  const int z  = blockIdx.z;             // w*3 + j
  const int k0 = blockIdx.x * 32;        // k within 2048-chunk
  const int n0 = blockIdx.y * 32;        // n within 1152
  const float* src = (z < 3) ? kw : vw;
  const int rowbase = (z % 3) * 2048 + k0;  // row within 6144 x 1152
  const int tx = threadIdx.x & 31;
  const int ty = threadIdx.x >> 5;       // 0..7
#pragma unroll
  for (int yy = 0; yy < 32; yy += 8) {   // coalesced read along n
    float v = src[(size_t)(rowbase + ty + yy) * D_OUT + n0 + tx];
    tile[ty + yy][tx] = f2bf(v);
  }
  __syncthreads();
#pragma unroll
  for (int yy = 0; yy < 32; yy += 8) {   // coalesced write along k
    Wt[(size_t)z * D_OUT * D_IN + (size_t)(n0 + ty + yy) * D_IN + k0 + tx] =
        tile[tx][ty + yy];
  }
}

// ---------------- kernel 1: feature partial GEMMs (NT) ---------------------
// z = w*3 + j : P[z] (1000 x 1152) = X (1000 x 2048) @ Wt[z]^T (Wt: 1152 x 2048)
__global__ __launch_bounds__(256) void gemm_feat_kernel(
    const unsigned short* __restrict__ X,
    const unsigned short* __restrict__ Wt,
    float* __restrict__ P) {
  __shared__ __align__(16) unsigned short ldsA[BM * LDP];
  __shared__ __align__(16) unsigned short ldsB[BN * LDP];
  const int tid  = threadIdx.x;
  const int row0 = blockIdx.x * BM;
  const int col0 = blockIdx.y * BN;
  const int z    = blockIdx.z;
  const int M = MROWS_F, K = D_IN;
  const unsigned short* Bp = Wt + (size_t)z * D_OUT * D_IN;  // n-major, ld = 2048
  float* Cp = P + (size_t)z * MROWS_F * D_OUT;
  const int wid = tid >> 5, lane = tid & 31;
  const int wm = (wid & 3) * 32, wn = (wid >> 2) * 32;
  v8f acc[2][2] = {};
  for (int k0 = 0; k0 < K; k0 += BK) {
    if (k0 + BK < K) {                    // hint next K tile (global_prefetch_b8)
      __builtin_prefetch(X  + (size_t)(row0 + (tid >> 3)) * D_IN + k0 + BK, 0, 1);
      __builtin_prefetch(Bp + (size_t)(col0 + (tid >> 3)) * D_IN + k0 + BK, 0, 1);
    }
    // A tile 128x32 (row-major, vector 8B loads)
#pragma unroll
    for (int p = 0; p < 4; ++p) {
      int gi = p * 256 + tid;
      int r = gi >> 3, c4 = (gi & 7) << 2;
      uint2 val; val.x = 0u; val.y = 0u;
      int gr = row0 + r;
      if (gr < M) val = *(const uint2*)(X + (size_t)gr * D_IN + k0 + c4);
      *(uint2*)(ldsA + r * LDP + c4) = val;
    }
    // B tile 64x32 (n-major rows, contiguous K): straight vector copy
#pragma unroll
    for (int p = 0; p < 2; ++p) {
      int gi = p * 256 + tid;
      int r = gi >> 3, c4 = (gi & 7) << 2;
      uint2 val = *(const uint2*)(Bp + (size_t)(col0 + r) * D_IN + k0 + c4);
      *(uint2*)(ldsB + r * LDP + c4) = val;
    }
    __syncthreads();
    v16bf a0 = lds_frag_a(ldsA, wm, lane);
    v16bf a1 = lds_frag_a(ldsA, wm + 16, lane);
    v16bf b0 = lds_frag_b(ldsB, wn, lane);
    v16bf b1 = lds_frag_b(ldsB, wn + 16, lane);
    acc[0][0] = wmma_bf16(a0, b0, acc[0][0]);
    acc[0][1] = wmma_bf16(a0, b1, acc[0][1]);
    acc[1][0] = wmma_bf16(a1, b0, acc[1][0]);
    acc[1][1] = wmma_bf16(a1, b1, acc[1][1]);
    __syncthreads();
  }
  const int moff = (lane < 16) ? 0 : 8;
#pragma unroll
  for (int i = 0; i < 2; ++i)
#pragma unroll
    for (int j = 0; j < 2; ++j) {
      AccU u; u.v = acc[i][j];
      int nn = col0 + wn + j * 16 + (lane & 15);
#pragma unroll
      for (int e = 0; e < 8; ++e) {
        int mm = row0 + wm + i * 16 + moff + e;
        if (mm < M) Cp[(size_t)mm * D_OUT + nn] = u.f[e];
      }
    }
}

// ---------------- kernel 2: tuple compose + bias + LayerNorm ---------------
__global__ __launch_bounds__(256) void compose_ln_kernel(
    const float* __restrict__ P,
    const float* __restrict__ kbias, const float* __restrict__ vbias,
    const float* __restrict__ g, const float* __restrict__ beta,
    unsigned short* __restrict__ sk_bf, unsigned short* __restrict__ qk_bf,
    unsigned short* __restrict__ sv_bf, float* __restrict__ qv) {
  const int row = blockIdx.x;                 // 0..6999 = n*56 + t
  const int n = row / T_LEN, t = row % T_LEN;
  const int t0 = TUP[t][0], t1 = TUP[t][1], t2 = TUP[t][2];
  __shared__ float kbuf[D_OUT];
  __shared__ float red[256];
  const float* Pk = P;
  const float* Pv = P + (size_t)3 * MROWS_F * D_OUT;
  const size_t b0 = ((size_t)(0 * MROWS_F + n * SEQ_LEN + t0)) * D_OUT;
  const size_t b1 = ((size_t)(1 * MROWS_F + n * SEQ_LEN + t1)) * D_OUT;
  const size_t b2 = ((size_t)(2 * MROWS_F + n * SEQ_LEN + t2)) * D_OUT;
  float sum = 0.f, ssq = 0.f;
  for (int d = threadIdx.x; d < D_OUT; d += 256) {
    float ck = kbias[d] + Pk[b0 + d] + Pk[b1 + d] + Pk[b2 + d];
    float cv = vbias[d] + Pv[b0 + d] + Pv[b1 + d] + Pv[b2 + d];
    kbuf[d] = ck;
    sum += ck; ssq += ck * ck;
    if (n < N_SUP) sv_bf[(size_t)row * D_OUT + d] = f2bf(cv);
    else           qv[(size_t)(row - N_SUP * T_LEN) * D_OUT + d] = cv;
  }
  red[threadIdx.x] = sum; __syncthreads();
  for (int s = 128; s > 0; s >>= 1) {
    if (threadIdx.x < s) red[threadIdx.x] += red[threadIdx.x + s];
    __syncthreads();
  }
  float tot = red[0]; __syncthreads();
  red[threadIdx.x] = ssq; __syncthreads();
  for (int s = 128; s > 0; s >>= 1) {
    if (threadIdx.x < s) red[threadIdx.x] += red[threadIdx.x + s];
    __syncthreads();
  }
  float tot2 = red[0];
  float mu = tot * (1.0f / D_OUT);
  float var = tot2 * (1.0f / D_OUT) - mu * mu;
  float rstd = rsqrtf(var + LN_EPS);
  for (int d = threadIdx.x; d < D_OUT; d += 256) {
    float v = (kbuf[d] - mu) * rstd * g[d] + beta[d];
    unsigned short h = f2bf(v);
    if (n < N_SUP) sk_bf[(size_t)row * D_OUT + d] = h;
    else           qk_bf[(size_t)(row - N_SUP * T_LEN) * D_OUT + d] = h;
  }
}

// ---------------- kernel 3: scores = qk . sk_c^T (NT GEMM, K=1152) ---------
__global__ __launch_bounds__(256) void gemm_scores_kernel(
    const unsigned short* __restrict__ QK,   // 5600 x 1152
    const unsigned short* __restrict__ SK,   // 25*56 x 1152
    float* __restrict__ S) {                 // 5 x 5600 x 280
  __shared__ __align__(16) unsigned short ldsA[BM * LDP];
  __shared__ __align__(16) unsigned short ldsB[BN * LDP];
  const int tid  = threadIdx.x;
  const int row0 = blockIdx.x * BM;
  const int col0 = blockIdx.y * BN;
  const int c    = blockIdx.z;
  const int M = MROWS_Q, K = D_OUT, N = KS;
  const unsigned short* Bp = SK + (size_t)c * KS * D_OUT;
  float* Cp = S + (size_t)c * MROWS_Q * KS;
  const int wid = tid >> 5, lane = tid & 31;
  const int wm = (wid & 3) * 32, wn = (wid >> 2) * 32;
  v8f acc[2][2] = {};
  for (int k0 = 0; k0 < K; k0 += BK) {
    if (k0 + BK < K) {                    // global_prefetch_b8 of next K tile
      __builtin_prefetch(QK + (size_t)(row0 + (tid >> 3)) * D_OUT + k0 + BK, 0, 1);
      if (col0 + (tid >> 3) < N)
        __builtin_prefetch(Bp + (size_t)(col0 + (tid >> 3)) * D_OUT + k0 + BK, 0, 1);
    }
#pragma unroll
    for (int p = 0; p < 4; ++p) {            // A tile 128x32
      int gi = p * 256 + tid;
      int r = gi >> 3, c4 = (gi & 7) << 2;
      uint2 val; val.x = 0u; val.y = 0u;
      int gr = row0 + r;
      if (gr < M) val = *(const uint2*)(QK + (size_t)gr * D_OUT + k0 + c4);
      *(uint2*)(ldsA + r * LDP + c4) = val;
    }
#pragma unroll
    for (int p = 0; p < 2; ++p) {            // B tile: rows of sk (N-major) 64x32
      int gi = p * 256 + tid;
      int r = gi >> 3, c4 = (gi & 7) << 2;
      uint2 val; val.x = 0u; val.y = 0u;
      int gn = col0 + r;
      if (gn < N) val = *(const uint2*)(Bp + (size_t)gn * D_OUT + k0 + c4);
      *(uint2*)(ldsB + r * LDP + c4) = val;
    }
    __syncthreads();
    v16bf a0 = lds_frag_a(ldsA, wm, lane);
    v16bf a1 = lds_frag_a(ldsA, wm + 16, lane);
    v16bf b0 = lds_frag_b(ldsB, wn, lane);
    v16bf b1 = lds_frag_b(ldsB, wn + 16, lane);
    acc[0][0] = wmma_bf16(a0, b0, acc[0][0]);
    acc[0][1] = wmma_bf16(a0, b1, acc[0][1]);
    acc[1][0] = wmma_bf16(a1, b0, acc[1][0]);
    acc[1][1] = wmma_bf16(a1, b1, acc[1][1]);
    __syncthreads();
  }
  const int moff = (lane < 16) ? 0 : 8;
#pragma unroll
  for (int i = 0; i < 2; ++i)
#pragma unroll
    for (int j = 0; j < 2; ++j) {
      AccU u; u.v = acc[i][j];
      int nn = col0 + wn + j * 16 + (lane & 15);
#pragma unroll
      for (int e = 0; e < 8; ++e) {
        int mm = row0 + wm + i * 16 + moff + e;
        if (mm < M && nn < N) Cp[(size_t)mm * KS + nn] = u.f[e] * SM_SCALE;
      }
    }
}

// ---------------- kernel 4: softmax over 280 (one wave per row) ------------
__global__ __launch_bounds__(256) void softmax_kernel(
    float* __restrict__ S, unsigned short* __restrict__ SMB) {
  const int wid = threadIdx.x >> 5, lane = threadIdx.x & 31;
  const int row = blockIdx.x * 8 + wid;
  if (row >= WAY * MROWS_Q) return;
  float* p = S + (size_t)row * KS;
  float v[9];
  float mx = -3.4e38f;
#pragma unroll
  for (int i = 0; i < 9; ++i) {
    int c = lane + i * 32;
    v[i] = (c < KS) ? p[c] : -3.4e38f;
    mx = fmaxf(mx, v[i]);
  }
#pragma unroll
  for (int o = 16; o > 0; o >>= 1) mx = fmaxf(mx, __shfl_xor(mx, o, 32));
  float s = 0.f;
#pragma unroll
  for (int i = 0; i < 9; ++i) {
    int c = lane + i * 32;
    float e = (c < KS) ? __expf(v[i] - mx) : 0.0f;
    v[i] = e; s += e;
  }
#pragma unroll
  for (int o = 16; o > 0; o >>= 1) s += __shfl_xor(s, o, 32);
  float inv = 1.0f / s;
#pragma unroll
  for (int i = 0; i < 9; ++i) {
    int c = lane + i * 32;
    if (c < KS) {
      float r = v[i] * inv;
      p[c] = r;
      SMB[(size_t)row * KS + c] = f2bf(r);
    }
  }
}

// ---------------- kernel 5: proto GEMM (NN, K=280) + fused distance --------
__global__ __launch_bounds__(256) void gemm_proto_kernel(
    const unsigned short* __restrict__ SM,  // 5 x 5600 x 280 bf16
    const unsigned short* __restrict__ SV,  // 25*56 x 1152 bf16
    const float* __restrict__ QV,           // 5600 x 1152
    float* __restrict__ ACC) {              // 5 x 100 partial sums
  __shared__ __align__(16) unsigned short ldsA[BM * LDP];
  __shared__ __align__(16) unsigned short ldsB[BN * LDP];
  __shared__ float rowsum[BM];
  const int tid  = threadIdx.x;
  const int row0 = blockIdx.x * BM;
  const int col0 = blockIdx.y * BN;
  const int c    = blockIdx.z;
  const int M = MROWS_Q, K = KS;
  const unsigned short* Ap = SM + (size_t)c * MROWS_Q * KS;
  const unsigned short* Bp = SV + (size_t)c * KS * D_OUT;
  const int wid = tid >> 5, lane = tid & 31;
  const int wm = (wid & 3) * 32, wn = (wid >> 2) * 32;
  v8f acc[2][2] = {};
  for (int k0 = 0; k0 < K; k0 += BK) {       // 9 iterations (288 padded)
#pragma unroll
    for (int p = 0; p < 4; ++p) {            // A tile 128x32, lda=280
      int gi = p * 256 + tid;
      int r = gi >> 3, c4 = (gi & 7) << 2;
      uint2 val; val.x = 0u; val.y = 0u;
      int gr = row0 + r;
      if (gr < M && (k0 + c4) < K)
        val = *(const uint2*)(Ap + (size_t)gr * KS + k0 + c4);
      *(uint2*)(ldsA + r * LDP + c4) = val;
    }
#pragma unroll
    for (int p = 0; p < 2; ++p) {            // B tile 32x64 row-major, transpose
      int gi = p * 256 + tid;
      int kr = gi >> 4, c4 = (gi & 15) << 2;
      U2 val; val.u.x = 0u; val.u.y = 0u;
      int gk = k0 + kr;
      if (gk < K) val.u = *(const uint2*)(Bp + (size_t)gk * D_OUT + col0 + c4);
#pragma unroll
      for (int e = 0; e < 4; ++e) ldsB[(c4 + e) * LDP + kr] = val.s[e];
    }
    __syncthreads();
    v16bf a0 = lds_frag_a(ldsA, wm, lane);
    v16bf a1 = lds_frag_a(ldsA, wm + 16, lane);
    v16bf b0 = lds_frag_b(ldsB, wn, lane);
    v16bf b1 = lds_frag_b(ldsB, wn + 16, lane);
    acc[0][0] = wmma_bf16(a0, b0, acc[0][0]);
    acc[0][1] = wmma_bf16(a0, b1, acc[0][1]);
    acc[1][0] = wmma_bf16(a1, b0, acc[1][0]);
    acc[1][1] = wmma_bf16(a1, b1, acc[1][1]);
    __syncthreads();
  }
  // fused epilogue: sum_{t,d in tile} (qv - proto)^2 per row -> per-q atomics
  if (tid < BM) rowsum[tid] = 0.0f;
  __syncthreads();
  const int moff = (lane < 16) ? 0 : 8;
#pragma unroll
  for (int i = 0; i < 2; ++i)
#pragma unroll
    for (int j = 0; j < 2; ++j) {
      AccU u; u.v = acc[i][j];
      int nn = col0 + wn + j * 16 + (lane & 15);
#pragma unroll
      for (int e = 0; e < 8; ++e) {
        int lr = wm + i * 16 + moff + e;     // local row 0..127
        int mm = row0 + lr;
        if (mm < M) {
          float diff = QV[(size_t)mm * D_OUT + nn] - u.f[e];
          atomicAdd(&rowsum[lr], diff * diff);
        }
      }
    }
  __syncthreads();
  if (tid < BM) {
    int mm = row0 + tid;
    if (mm < M) {
      int q = mm / T_LEN;
      atomicAdd(&ACC[c * N_Q + q], rowsum[tid]);
    }
  }
}

// ---------------- kernel 6: zero accumulator -------------------------------
__global__ __launch_bounds__(256) void zero_acc_kernel(float* __restrict__ a) {
  int i = blockIdx.x * 256 + threadIdx.x;
  if (i < WAY * N_Q) a[i] = 0.0f;
}

// ---------------- kernel 7: logits + align outputs -------------------------
__global__ __launch_bounds__(256) void finalize_kernel(
    const float* __restrict__ ACC, const float* __restrict__ S4,
    float* __restrict__ out) {
  int i = blockIdx.x * 256 + threadIdx.x;
  if (i < N_Q * WAY) {
    int q = i / WAY, c = i % WAY;
    out[i] = -ACC[c * N_Q + q] * (1.0f / (float)T_LEN);
  }
  int j = i - N_Q * WAY;
  if (j >= 0 && j < N_Q * T_LEN * T_LEN) {
    int qt = j / T_LEN, s2 = j % T_LEN;      // qt = q*56 + t
    float s = 0.f;
#pragma unroll
    for (int k = 0; k < SHOT; ++k) s += S4[(size_t)qt * KS + k * T_LEN + s2];
    out[N_Q * WAY + j] = s;
  }
}

// ---------------------------------------------------------------------------
extern "C" void kernel_launch(void* const* d_in, const int* in_sizes, int n_in,
                              void* d_out, int out_size, void* d_ws, size_t ws_size,
                              hipStream_t stream) {
  const float* sup  = (const float*)d_in[0];
  const float* qry  = (const float*)d_in[1];
  // d_in[2] = support_labels (unused; class blocks are implicit)
  const float* kw   = (const float*)d_in[3];
  const float* kb   = (const float*)d_in[4];
  const float* vw   = (const float*)d_in[5];
  const float* vb   = (const float*)d_in[6];
  const float* lng  = (const float*)d_in[7];
  const float* lnb  = (const float*)d_in[8];

  char* ws = (char*)d_ws;
  // Region A [0, 28311552): bf16 transposed weights (prep_wt -> gemm_feat),
  // then qv (fp32) once weights are dead.
  unsigned short* Wt  = (unsigned short*)(ws + 0);
  float* qv           = (float*)(ws + 0);                 // 25,804,800 B
  // Region B [28311552, 60055552): Xbf + P, then scores (fp32)
  unsigned short* Xbf = (unsigned short*)(ws + 28311552); //  4,096,000 B
  float* P            = (float*)(ws + 32407552);          // 27,648,000 B
  float* scores       = (float*)(ws + 28311552);          // 31,360,000 B (alias)
  // Region C: attention operands + accumulator
  unsigned short* qk_bf = (unsigned short*)(ws + 60055552); // 12,902,400 B
  unsigned short* sk_bf = (unsigned short*)(ws + 72957952); //  3,225,600 B
  unsigned short* sv_bf = (unsigned short*)(ws + 76183552); //  3,225,600 B
  unsigned short* sm_bf = (unsigned short*)(ws + 79409152); // 15,680,000 B
  float* accg           = (float*)(ws + 95089152);          //      2,000 B
  // total ws requirement: ~95.1 MB

  // 1) positional encoding + bf16 cast of activations
  prep_x_kernel<<<(N_ALL * SEQ_LEN * D_IN + 255) / 256, 256, 0, stream>>>(sup, qry, Xbf);
  // 2) weights -> bf16, transposed to n-major (per 2048-row chunk)
  prep_wt_kernel<<<dim3(64, 36, 6), 256, 0, stream>>>(kw, vw, Wt);
  // 3) feature partial GEMMs: P[w*3+j] = X @ Wchunk, 6 x (1000 x 2048 x 1152)
  gemm_feat_kernel<<<dim3(8, 18, 6), 256, 0, stream>>>(Xbf, Wt, P);
  // 4) tuple compose + bias + LayerNorm -> sk/qk (bf16), sv (bf16), qv (fp32)
  compose_ln_kernel<<<N_ALL * T_LEN, 256, 0, stream>>>(P, kb, vb, lng, lnb,
                                                       sk_bf, qk_bf, sv_bf, qv);
  // 5) scores GEMM (NT), per class, scaled by 1/sqrt(D_OUT)
  gemm_scores_kernel<<<dim3(44, 5, WAY), 256, 0, stream>>>(qk_bf, sk_bf, scores);
  // 6) softmax over 280, fp32 in place + bf16 copy for proto GEMM
  softmax_kernel<<<(WAY * MROWS_Q) / 8, 256, 0, stream>>>(scores, sm_bf);
  // 7) zero distance accumulators (ws is poisoned, must re-zero every call)
  zero_acc_kernel<<<2, 256, 0, stream>>>(accg);
  // 8) proto GEMM (NN) with fused (qv - proto)^2 reduction
  gemm_proto_kernel<<<dim3(44, 18, WAY), 256, 0, stream>>>(sm_bf, sv_bf, qv, accg);
  // 9) logits + align (align comes from last class c=4 softmax)
  finalize_kernel<<<(N_Q * WAY + N_Q * T_LEN * T_LEN + 255) / 256, 256, 0, stream>>>(
      accg, scores + (size_t)4 * MROWS_Q * KS, (float*)d_out);
}
